// RuleNetwork_58153857188098
// MI455X (gfx1250) — matrix operand order repaired and checked by reading, not compiled, not run on hardware
//
#include <hip/hip_runtime.h>
#include <hip/hip_bf16.h>
#include <cstddef>

// ---------------------------------------------------------------------------
// Problem constants (from the reference)
// ---------------------------------------------------------------------------
constexpr int Hd   = 512;      // hidden dim
constexpr int NV   = 6;        // num variables
constexpr int NR   = 4;        // num rules
constexpr int Bb   = 4096;     // batch
constexpr int NH   = 4;        // heads
constexpr int FF   = 2048;     // ff dim
constexpr int SEQ  = NV + NR;  // 10
constexpr int DK   = Hd / NH;  // 128
constexpr int Mrows = SEQ * Bb;        // 40960 GEMM rows
constexpr float INV_SQRT_DK = 0.08838834764831845f; // 1/sqrt(128)

typedef __bf16 bf16;
typedef __attribute__((ext_vector_type(8)))  bf16  v8bf;
typedef __attribute__((ext_vector_type(16))) bf16  v16bf;
typedef __attribute__((ext_vector_type(8)))  float v8f;

// ---------------------------------------------------------------------------
// f32 -> bf16 conversion (weights), grid-stride
// ---------------------------------------------------------------------------
__global__ void cvt_f32_bf16(const float* __restrict__ in, bf16* __restrict__ out,
                             size_t n) {
  size_t i = (size_t)blockIdx.x * blockDim.x + threadIdx.x;
  size_t stride = (size_t)gridDim.x * blockDim.x;
  for (; i < n; i += stride) out[i] = (bf16)in[i];
}

// ---------------------------------------------------------------------------
// Build x = concat(hidden, rule_emb).transpose(1,0,2) + PE ; also bf16 copy
// x layout: [s][b][h], row index = s*Bb + b
// ---------------------------------------------------------------------------
__global__ void build_x_kernel(const float* __restrict__ hidden,
                               const float* __restrict__ rule_emb,
                               float* __restrict__ x, bf16* __restrict__ xb) {
  size_t i = (size_t)blockIdx.x * blockDim.x + threadIdx.x;
  if (i >= (size_t)SEQ * Bb * Hd) return;
  int hh = (int)(i % Hd);
  size_t sb = i / Hd;
  int b = (int)(sb % Bb);
  int s = (int)(sb / Bb);
  float v = (s < NV) ? hidden[((size_t)b * NV + s) * Hd + hh]
                     : rule_emb[(size_t)(s - NV) * Hd + hh];
  int i2 = hh & ~1;
  float dv = __expf(-(float)i2 * (9.210340371976184f / (float)Hd)); // ln(10000)
  float pe = (hh & 1) ? __cosf((float)s * dv) : __sinf((float)s * dv);
  v += pe;
  x[i] = v;
  xb[i] = (bf16)v;
}

// ---------------------------------------------------------------------------
// WMMA bf16 GEMM:  C[M,N] = A[M,K] * W[N,K]^T + bias, compile-time RELU and
// output type (f32 or bf16) so the epilogue is branch-free straight-line code.
// Block: 256 thr = 8 wave32. C tile 128x128; each wave owns 32x64
// (2x4 v_wmma_f32_16x16x32_bf16 tiles -> 8 WMMA per K slab).
// K staged in 32-wide slabs through double-buffered LDS; global->register->LDS
// staging with statically assigned chunks so loads pipeline. Stride-40 LDS
// rows keep 16B alignment for ds_load_b128 and avoid bank conflicts.
// ---------------------------------------------------------------------------
constexpr int LSTR = 40; // 32 halfs + pad (80B, 16B aligned)

template <int RELU, int OUT_BF16>
__global__ __launch_bounds__(256)
void gemm_bf16_wmma(const bf16* __restrict__ A, const bf16* __restrict__ W,
                    const float* __restrict__ bias, void* __restrict__ Cout,
                    int K, int ldC) {
  __shared__ bf16 As[2][128 * LSTR];
  __shared__ bf16 Ws[2][128 * LSTR];

  const int tid  = threadIdx.x;
  const int lane = tid & 31;
  const int wid  = tid >> 5;
  const int wm   = (wid >> 1) * 32;   // wave row base in tile (0,32,64,96)
  const int wn   = (wid & 1) * 64;    // wave col base in tile (0,64)
  const int bm   = blockIdx.y * 128;
  const int bn   = blockIdx.x * 128;
  const int lane16  = lane & 15;
  const int halfsel = lane >> 4;      // 0: lanes 0-15, 1: lanes 16-31

  // Static staging assignment: thread owns row rA, 16 contiguous halfs at cA,
  // in both the A tile (128x32) and the W tile (128x32).
  const int rA = tid >> 1;
  const int cA = (tid & 1) * 16;
  const bf16* aptr = A + (size_t)(bm + rA) * K + cA;
  const bf16* wptr = W + (size_t)(bn + rA) * K + cA;
  const int lds_off = rA * LSTR + cA;

  // First slab: global -> regs -> LDS buffer 0
  v8bf ra0 = *(const v8bf*)aptr;
  v8bf ra1 = *(const v8bf*)(aptr + 8);
  v8bf rw0 = *(const v8bf*)wptr;
  v8bf rw1 = *(const v8bf*)(wptr + 8);
  *(v8bf*)&As[0][lds_off]     = ra0;
  *(v8bf*)&As[0][lds_off + 8] = ra1;
  *(v8bf*)&Ws[0][lds_off]     = rw0;
  *(v8bf*)&Ws[0][lds_off + 8] = rw1;

  v8f acc[2][4] = {};
  int cur = 0;

  for (int k0 = 0; k0 < K; k0 += 32) {
    __syncthreads();                       // buf[cur] visible to all waves
    const bool more = (k0 + 32) < K;
    if (more) {                            // issue next slab's global loads now
      aptr += 32; wptr += 32;
      ra0 = *(const v8bf*)aptr;
      ra1 = *(const v8bf*)(aptr + 8);
      rw0 = *(const v8bf*)wptr;
      rw1 = *(const v8bf*)(wptr + 8);
      __builtin_prefetch(aptr + 32, 0, 1); // global_prefetch_b8: slab after next
      __builtin_prefetch(wptr + 32, 0, 1);
    }

    // A fragments (16x32, §7.12.2): lane holds row m=lane%16;
    // lanes 0-15: el0-7=K0..7, el8-15=K16..23 ; lanes16-31: K8..15 / K24..31
    v16bf afrag[2];
#pragma unroll
    for (int r = 0; r < 2; ++r) {
      int row = wm + r * 16 + lane16;
      v8bf lo = *(const v8bf*)&As[cur][row * LSTR + halfsel * 8];
      v8bf hi = *(const v8bf*)&As[cur][row * LSTR + 16 + halfsel * 8];
      afrag[r] = __builtin_shufflevector(lo, hi, 0, 1, 2, 3, 4, 5, 6, 7,
                                         8, 9, 10, 11, 12, 13, 14, 15);
    }
    // B fragments (32x16): lane holds col n=lane%16;
    // lanes 0-15: el0-7=K0..7, el8-15=K16..23 ; lanes16-31: K8..15 / K24..31
    v16bf wfrag[4];
#pragma unroll
    for (int t = 0; t < 4; ++t) {
      int row = wn + t * 16 + lane16;
      v8bf lo = *(const v8bf*)&Ws[cur][row * LSTR + halfsel * 8];
      v8bf hi = *(const v8bf*)&Ws[cur][row * LSTR + 16 + halfsel * 8];
      wfrag[t] = __builtin_shufflevector(lo, hi, 0, 1, 2, 3, 4, 5, 6, 7,
                                         8, 9, 10, 11, 12, 13, 14, 15);
    }
#pragma unroll
    for (int r = 0; r < 2; ++r)
#pragma unroll
      for (int t = 0; t < 4; ++t)
        acc[r][t] = __builtin_amdgcn_wmma_f32_16x16x32_bf16(
            false, afrag[r], false, wfrag[t], (short)0, acc[r][t], false, false);

    if (more) {                            // drain regs into the other buffer
      int nxt = cur ^ 1;
      *(v8bf*)&As[nxt][lds_off]     = ra0;
      *(v8bf*)&As[nxt][lds_off + 8] = ra1;
      *(v8bf*)&Ws[nxt][lds_off]     = rw0;
      *(v8bf*)&Ws[nxt][lds_off + 8] = rw1;
    }
    cur ^= 1;
  }

  // Epilogue per §7.12.2 C layout: VGPR v: lanes0-15 -> (M=v, N=lane16),
  // lanes16-31 -> (M=v+8, N=lane16). Branch-free: RELU/OUT_BF16 are
  // compile-time, bias is always present.
  float* Cf = (float*)Cout;
  bf16*  Cb = (bf16*)Cout;
  const int mo = halfsel * 8;
#pragma unroll
  for (int r = 0; r < 2; ++r) {
#pragma unroll
    for (int t = 0; t < 4; ++t) {
      int nidx = bn + wn + t * 16 + lane16;
      float bv = bias[nidx];
#pragma unroll
      for (int v = 0; v < 8; ++v) {
        int m = bm + wm + r * 16 + mo + v;
        float val = acc[r][t][v] + bv;
        if (RELU) val = fmaxf(val, 0.f);
        size_t idx = (size_t)m * ldC + nidx;
        if (OUT_BF16) Cb[idx] = (bf16)val;
        else          Cf[idx] = val;
      }
    }
  }
}

// ---------------------------------------------------------------------------
// Per-(batch,head) attention: S=10 so scores are tiny -> LDS/VALU kernel.
// qkv layout: [(s*Bb+b)*1536 + {0:q,512:k,1024:v} + head*128 + d], bf16.
// Writes attention output (bf16) at [(s*Bb+b)*512 + head*128 + d].
// ---------------------------------------------------------------------------
__global__ __launch_bounds__(128)
void attn_kernel(const bf16* __restrict__ qkv, bf16* __restrict__ out) {
  __shared__ float qs[SEQ][DK], ks[SEQ][DK], vs[SEQ][DK];
  __shared__ float sc[SEQ][SEQ];
  int b = blockIdx.x / NH;
  int h = blockIdx.x % NH;
  int t = threadIdx.x;
  for (int s = 0; s < SEQ; ++s) {
    size_t base = ((size_t)s * Bb + b) * (3 * Hd) + h * DK + t;
    qs[s][t] = (float)qkv[base];
    ks[s][t] = (float)qkv[base + Hd];
    vs[s][t] = (float)qkv[base + 2 * Hd];
  }
  __syncthreads();
  if (t < SEQ * SEQ) {
    int i = t / SEQ, j = t % SEQ;
    float d = 0.f;
    for (int e = 0; e < DK; ++e) d += qs[i][e] * ks[j][e];
    sc[i][j] = d * INV_SQRT_DK;
  }
  __syncthreads();
  if (t < SEQ) {
    float mx = -1e30f;
    for (int j = 0; j < SEQ; ++j) mx = fmaxf(mx, sc[t][j]);
    float sum = 0.f;
    for (int j = 0; j < SEQ; ++j) { float e = __expf(sc[t][j] - mx); sc[t][j] = e; sum += e; }
    float inv = 1.f / sum;
    for (int j = 0; j < SEQ; ++j) sc[t][j] *= inv;
  }
  __syncthreads();
  for (int s = 0; s < SEQ; ++s) {
    float o = 0.f;
    for (int j = 0; j < SEQ; ++j) o += sc[s][j] * vs[j][t];
    out[((size_t)s * Bb + b) * Hd + h * DK + t] = (bf16)o;
  }
}

// ---------------------------------------------------------------------------
// Fused residual + LayerNorm over H=512; updates x (f32) and xb (bf16).
// ---------------------------------------------------------------------------
__global__ __launch_bounds__(256)
void add_ln_kernel(float* __restrict__ x, const float* __restrict__ a,
                   const float* __restrict__ g, const float* __restrict__ bta,
                   bf16* __restrict__ xb) {
  __shared__ float red[256];
  size_t row = blockIdx.x;
  int t = threadIdx.x;
  float v0 = x[row * Hd + t]       + a[row * Hd + t];
  float v1 = x[row * Hd + t + 256] + a[row * Hd + t + 256];
  red[t] = v0 + v1;
  __syncthreads();
  for (int o = 128; o > 0; o >>= 1) { if (t < o) red[t] += red[t + o]; __syncthreads(); }
  float mean = red[0] * (1.f / Hd);
  __syncthreads();
  float d0 = v0 - mean, d1 = v1 - mean;
  red[t] = d0 * d0 + d1 * d1;
  __syncthreads();
  for (int o = 128; o > 0; o >>= 1) { if (t < o) red[t] += red[t + o]; __syncthreads(); }
  float inv = rsqrtf(red[0] * (1.f / Hd) + 1e-5f);
  float y0 = d0 * inv * g[t]       + bta[t];
  float y1 = d1 * inv * g[t + 256] + bta[t + 256];
  x[row * Hd + t] = y0;       x[row * Hd + t + 256] = y1;
  xb[row * Hd + t] = (bf16)y0; xb[row * Hd + t + 256] = (bf16)y1;
}

// ---------------------------------------------------------------------------
// Final MHA attention weights, averaged over heads: wm[b][10][10].
// Only q,k used (qkv buffer columns 0..1023).
// ---------------------------------------------------------------------------
__global__ __launch_bounds__(128)
void final_attn_kernel(const bf16* __restrict__ qkv, float* __restrict__ wm) {
  __shared__ float qs[SEQ][Hd], ks[SEQ][Hd];
  __shared__ float sc[SEQ][SEQ];
  __shared__ float accw[SEQ][SEQ];
  int b = blockIdx.x, t = threadIdx.x;
  for (int s = 0; s < SEQ; ++s)
    for (int j = 0; j < 4; ++j) {
      int col = t + j * 128;
      size_t base = ((size_t)s * Bb + b) * (3 * Hd);
      qs[s][col] = (float)qkv[base + col];
      ks[s][col] = (float)qkv[base + Hd + col];
    }
  if (t < SEQ * SEQ) accw[t / SEQ][t % SEQ] = 0.f;
  __syncthreads();
  for (int h = 0; h < NH; ++h) {
    if (t < SEQ * SEQ) {
      int i = t / SEQ, j = t % SEQ;
      float d = 0.f;
      for (int e = 0; e < DK; ++e) d += qs[i][h * DK + e] * ks[j][h * DK + e];
      sc[i][j] = d * INV_SQRT_DK;
    }
    __syncthreads();
    if (t < SEQ) {
      float mx = -1e30f;
      for (int j = 0; j < SEQ; ++j) mx = fmaxf(mx, sc[t][j]);
      float p[SEQ]; float sum = 0.f;
      for (int j = 0; j < SEQ; ++j) { p[j] = __expf(sc[t][j] - mx); sum += p[j]; }
      float inv = (1.f / NH) / sum;
      for (int j = 0; j < SEQ; ++j) accw[t][j] += p[j] * inv;
    }
    __syncthreads();
  }
  if (t < SEQ * SEQ) wm[(size_t)b * (SEQ * SEQ) + t] = accw[t / SEQ][t % SEQ];
}

// ---------------------------------------------------------------------------
// scores = vr + rv; argmax one-hot mask; out = hidden + scores*value
// ---------------------------------------------------------------------------
__global__ __launch_bounds__(256)
void final_out_kernel(const float* __restrict__ wm, const float* __restrict__ x,
                      const float* __restrict__ hidden, float* __restrict__ out) {
  __shared__ float sc[NV * NR];
  __shared__ int bi;
  __shared__ float bv;
  int b = blockIdx.x, t = threadIdx.x;
  if (t < NV * NR) {
    int v = t / NR, r = t % NR;
    const float* w = wm + (size_t)b * (SEQ * SEQ);
    sc[t] = w[v * SEQ + (NV + r)] + w[(NV + r) * SEQ + v];
  }
  __syncthreads();
  if (t == 0) {
    int best = 0; float m = sc[0];
    for (int k = 1; k < NV * NR; ++k)
      if (sc[k] > m) { m = sc[k]; best = k; }  // first max wins, like argmax
    bi = best; bv = m;
  }
  __syncthreads();
  int vb = bi / NR, rb = bi % NR;
  for (int e = t; e < NV * Hd; e += 256) {
    int v = e / Hd, hh = e % Hd;
    float o = hidden[((size_t)b * NV + v) * Hd + hh];
    if (v == vb) o += bv * x[(((size_t)(NV + rb) * Bb) + b) * Hd + hh];
    out[((size_t)b * NV + v) * Hd + hh] = o;
  }
}

// ---------------------------------------------------------------------------
// Host launcher
// ---------------------------------------------------------------------------
extern "C" void kernel_launch(void* const* d_in, const int* in_sizes, int n_in,
                              void* d_out, int out_size, void* d_ws, size_t ws_size,
                              hipStream_t stream) {
  (void)in_sizes; (void)n_in; (void)out_size; (void)ws_size;
  const float* hidden   = (const float*)d_in[0];
  const float* rule_emb = (const float*)d_in[1];
  const float* t_in_w   = (const float*)d_in[2];
  const float* t_in_b   = (const float*)d_in[3];
  const float* t_out_w  = (const float*)d_in[4];
  const float* t_out_b  = (const float*)d_in[5];
  const float* t_ff1_w  = (const float*)d_in[6];
  const float* t_ff1_b  = (const float*)d_in[7];
  const float* t_ff2_w  = (const float*)d_in[8];
  const float* t_ff2_b  = (const float*)d_in[9];
  const float* t_ln1_g  = (const float*)d_in[10];
  const float* t_ln1_b  = (const float*)d_in[11];
  const float* t_ln2_g  = (const float*)d_in[12];
  const float* t_ln2_b  = (const float*)d_in[13];
  const float* mha_in_w = (const float*)d_in[14];
  const float* mha_in_b = (const float*)d_in[15];
  float* out = (float*)d_out;

  char* ws = (char*)d_ws;
  size_t off = 0;
  auto take = [&](size_t bytes) { char* p = ws + off; off += (bytes + 255) & ~(size_t)255; return p; };

  float* xf   = (float*)take((size_t)Mrows * Hd * 4);        // residual stream f32
  float* aout = (float*)take((size_t)Mrows * Hd * 4);        // sublayer output f32
  bf16*  xb   = (bf16*) take((size_t)Mrows * Hd * 2);        // x in bf16 (GEMM A)
  bf16*  qkvb = (bf16*) take((size_t)Mrows * 3 * Hd * 2);    // qkv bf16
  bf16*  atnb = (bf16*) take((size_t)Mrows * Hd * 2);        // attn output bf16
  bf16*  fb   = (bf16*) take((size_t)Mrows * FF * 2);        // ff hidden bf16
  float* wm   = (float*)take((size_t)Bb * SEQ * SEQ * 4);    // mean attn weights

  bf16* w_in[3];  bf16* w_out[3]; bf16* w_f1[3]; bf16* w_f2[3];
  for (int i = 0; i < 3; ++i) {
    w_in[i]  = (bf16*)take((size_t)3 * Hd * Hd * 2);   // 1536 x 512
    w_out[i] = (bf16*)take((size_t)Hd * Hd * 2);       // 512 x 512
    w_f1[i]  = (bf16*)take((size_t)FF * Hd * 2);       // 2048 x 512
    w_f2[i]  = (bf16*)take((size_t)Hd * FF * 2);       // 512 x 2048
  }
  bf16* w_mha = (bf16*)take((size_t)3 * Hd * Hd * 2);

  auto cvt = [&](const float* src, bf16* dst, size_t n) {
    cvt_f32_bf16<<<1024, 256, 0, stream>>>(src, dst, n);
  };
  for (int i = 0; i < 3; ++i) {
    cvt(t_in_w  + (size_t)i * 3 * Hd * Hd, w_in[i],  (size_t)3 * Hd * Hd);
    cvt(t_out_w + (size_t)i * Hd * Hd,     w_out[i], (size_t)Hd * Hd);
    cvt(t_ff1_w + (size_t)i * FF * Hd,     w_f1[i],  (size_t)FF * Hd);
    cvt(t_ff2_w + (size_t)i * Hd * FF,     w_f2[i],  (size_t)Hd * FF);
  }
  cvt(mha_in_w, w_mha, (size_t)3 * Hd * Hd);

  {
    size_t total = (size_t)SEQ * Bb * Hd;
    build_x_kernel<<<(int)((total + 255) / 256), 256, 0, stream>>>(hidden, rule_emb, xf, xb);
  }

  // GEMM dispatchers (compile-time epilogue variants)
  auto gemm_f32 = [&](const bf16* A, const bf16* W, const float* bias,
                      float* C, int N, int K, int ldC) {
    dim3 grid(N / 128, Mrows / 128);
    gemm_bf16_wmma<0, 0><<<grid, 256, 0, stream>>>(A, W, bias, (void*)C, K, ldC);
  };
  auto gemm_bf = [&](const bf16* A, const bf16* W, const float* bias,
                     bf16* C, int N, int K, int ldC) {
    dim3 grid(N / 128, Mrows / 128);
    gemm_bf16_wmma<0, 1><<<grid, 256, 0, stream>>>(A, W, bias, (void*)C, K, ldC);
  };
  auto gemm_bf_relu = [&](const bf16* A, const bf16* W, const float* bias,
                          bf16* C, int N, int K, int ldC) {
    dim3 grid(N / 128, Mrows / 128);
    gemm_bf16_wmma<1, 1><<<grid, 256, 0, stream>>>(A, W, bias, (void*)C, K, ldC);
  };

  // --- 3 transformer layers ---
  for (int i = 0; i < 3; ++i) {
    gemm_bf(xb, w_in[i], t_in_b + (size_t)i * 3 * Hd, qkvb, 3 * Hd, Hd, 3 * Hd);
    attn_kernel<<<Bb * NH, 128, 0, stream>>>(qkvb, atnb);
    gemm_f32(atnb, w_out[i], t_out_b + (size_t)i * Hd, aout, Hd, Hd, Hd);
    add_ln_kernel<<<Mrows, 256, 0, stream>>>(xf, aout,
        t_ln1_g + (size_t)i * Hd, t_ln1_b + (size_t)i * Hd, xb);
    gemm_bf_relu(xb, w_f1[i], t_ff1_b + (size_t)i * FF, fb, FF, Hd, FF);
    gemm_f32(fb, w_f2[i], t_ff2_b + (size_t)i * Hd, aout, Hd, FF, Hd);
    add_ln_kernel<<<Mrows, 256, 0, stream>>>(xf, aout,
        t_ln2_g + (size_t)i * Hd, t_ln2_b + (size_t)i * Hd, xb);
  }

  // --- final MHA: only q,k needed -> compute first 1024 columns of qkv ---
  gemm_bf(xb, w_mha, mha_in_b, qkvb, 2 * Hd, Hd, 3 * Hd);
  final_attn_kernel<<<Bb, 128, 0, stream>>>(qkvb, wm);

  // --- scores / argmax mask / einsum epilogue ---
  final_out_kernel<<<Bb, 256, 0, stream>>>(wm, xf, hidden, out);
}
